// DeTGC_30511447670797
// MI455X (gfx1250) — matrix operand order repaired and checked by reading, not compiled, not run on hardware
//
#include <hip/hip_runtime.h>

// DeTGC on gfx1250 (MI455X).
// GEMM view: M=O=128, N=N*T*V=102400, K=C*eta=640 (5 accumulating GEMMs, K=128).
// Block = 256 threads = 8 wave32; each wave owns 32 columns x all 128 rows
// (16 v8f accumulators). 400 blocks cover all columns exactly.
// Weight slice W[:,:,e] staged in LDS as f16 via CDNA5 TDM (tensor_load_to_lds)
// from a pre-converted f16 [e][o][c] buffer in d_ws (VALU staging fallback if
// the workspace is too small).
// B-fragments: 64 L2-resident gathers per k-chunk are batched into distinct
// registers and issued one k-chunk ahead so latency hides under the WMMA sweep.

typedef __attribute__((ext_vector_type(16))) _Float16     v16h;
typedef __attribute__((ext_vector_type(8)))  float        v8f;
typedef __attribute__((ext_vector_type(4)))  unsigned int v4u;
typedef __attribute__((ext_vector_type(8)))  int          v8i;
typedef __attribute__((ext_vector_type(4)))  int          v4i;

namespace {
constexpr int kN = 64, kC = 128, kT = 64, kV = 25, kEta = 5, kO = 128;
constexpr int kColsPerN  = kT * kV;         // 1600
constexpr int kCols      = kN * kColsPerN;  // 102400
constexpr int kColsPerWG = 256;             // 8 waves * 32 cols
constexpr int kWElems    = kO * kC * kEta;  // 81920
}

union AFrag { v16h v; uint4 q[2]; };
union BFrag { v16h v; _Float16 h[16]; };

// One-time: conv_w (o,c,e) f32 -> d_ws as f16 [e][o][c] (32KB per e-slice).
__global__ __launch_bounds__(256) void convert_w_f16(
    const float* __restrict__ cw, _Float16* __restrict__ wf) {
  int i = blockIdx.x * 256 + threadIdx.x;
  if (i < kWElems) {
    int o = i / (kC * kEta);
    int r = i - o * (kC * kEta);
    int c = r / kEta;
    int e = r - c * kEta;
    wf[(e * kO + o) * kC + c] = (_Float16)cw[i];
  }
}

__global__ __launch_bounds__(256, 2) void detgc_wmma_f16(
    const float* __restrict__ x,  const float* __restrict__ tr,
    const float* __restrict__ cw, const float* __restrict__ cb,
    const float* __restrict__ g,  const float* __restrict__ be,
    const float* __restrict__ mu, const float* __restrict__ va,
    const _Float16* __restrict__ wf16, int use_tdm,
    float* __restrict__ out)
{
  __shared__ __align__(16) _Float16 Wlds[kO * kC];  // 32 KB e-slice, [o][c]
  __shared__ float sA[kO];
  __shared__ float sB[kO];

  const int tid  = threadIdx.x;
  const int lane = tid & 31;
  const int wave = tid >> 5;

  if (tid < kO) {
    float inv = g[tid] * rsqrtf(va[tid] + 1e-5f);
    sA[tid] = inv;
    sB[tid] = cb[tid] * inv + be[tid] - mu[tid] * inv;
  }

  // Two column fragments per wave: cols [base, base+16) and [base+16, base+32).
  // kColsPerN (1600) is a multiple of 32, so both share the same n.
  const int colBase = blockIdx.x * kColsPerWG + wave * 32;
  const int col0 = colBase + (lane & 15);
  const int col1 = col0 + 16;
  const int n    = col0 / kColsPerN;
  const int rem0 = col0 % kColsPerN;  const int rem1 = col1 % kColsPerN;
  const int t0   = rem0 / kV;         const int t1   = rem1 / kV;
  const int v0   = rem0 - t0 * kV;    const int v1   = rem1 - t1 * kV;
  const float* xn = x + (size_t)n * (kC * kColsPerN);

  // WMMA f16 lane mappings used here (wave32, contiguous k; not the ISA table order):
  //  B 32x16: lane%16 = N; lanes<16 hold K=h, lanes>=16 hold K=16+h
  //  A 16x32: lane%16 = M; lanes<16 hold K=h, lanes>=16 hold K=16+h
  //           (one contiguous 16-half run, matching B).
  const int koffB = (lane < 16) ? 0 : 16;
  const int koffA = (lane < 16) ? 0 : 16;
  const int mrow  = lane & 15;

  v8f acc[16] = {};

  for (int e = 0; e < kEta; ++e) {
    __syncthreads();  // everyone done reading previous Wlds slice
    if (use_tdm) {
      if (wave == 0) {
        // Tensor DMA: copy 32KB f16 slice d_ws[e] -> LDS (one 1-line tile of
        // 4096 x 8-byte elements). D# per ISA ch.8; groups 2/3 unused (zero).
        unsigned long long ga =
            (unsigned long long)(uintptr_t)(wf16 + (size_t)e * kO * kC);
        unsigned lds_off = (unsigned)(uintptr_t)(void*)&Wlds[0];
        v4u g0;
        g0[0] = 1u;                                   // count=1, user mode
        g0[1] = lds_off;                              // lds_addr
        g0[2] = (unsigned)(ga & 0xFFFFFFFFu);         // global_addr[31:0]
        g0[3] = (unsigned)((ga >> 32) & 0x01FFFFFFu)  // global_addr[56:32]
                | (2u << 30);                         // type=2 (image)
        v8i g1;
        g1[0] = (int)(3u << 16);          // data_size=3 (8B), no mask/pad
        g1[1] = (int)(4096u << 16);       // tensor_dim0[15:0]=4096
        g1[2] = (int)(1u << 16);          // tensor_dim0[31:16]=0, tensor_dim1=1
        g1[3] = (int)(4096u << 16);       // tile_dim0=4096
        g1[4] = 1;                        // tile_dim1=1, tile_dim2=0
        g1[5] = 4096;                     // tensor_dim0_stride[31:0]
        g1[6] = 0;
        g1[7] = 0;
        v4i gz4 = {0, 0, 0, 0};
        v8i gz8 = {0, 0, 0, 0, 0, 0, 0, 0};
        __builtin_amdgcn_tensor_load_to_lds(g0, g1, gz4, gz4, gz8, 0);
        __builtin_amdgcn_s_wait_tensorcnt(0);
      }
    } else {
      for (int i = tid; i < kO * kC; i += 256)
        Wlds[i] = (_Float16)cw[i * kEta + e];
    }
    __syncthreads();

    // Interpolation params (matches ref: alpha = clip(ceil(t)-t,0,1),
    // xi = x1*(1-alpha) + x2*alpha) for both column fragments.
    float tr_e = tr[e];
    float tf0 = fminf(fmaxf((float)t0 + tr_e, 0.0f), (float)(kT - 1));
    float tf1 = fminf(fmaxf((float)t1 + tr_e, 0.0f), (float)(kT - 1));
    float a20 = fminf(fmaxf(ceilf(tf0) - tf0, 0.0f), 1.0f);
    float a21 = fminf(fmaxf(ceilf(tf1) - tf1, 0.0f), 1.0f);
    float a10 = 1.0f - a20, a11 = 1.0f - a21;
    // Fold the lane's K-offset (koffB) into the gather bases.
    const float* q10 = xn + (int)floorf(tf0) * kV + v0 + koffB * kColsPerN;
    const float* q20 = xn + (int)ceilf(tf0)  * kV + v0 + koffB * kColsPerN;
    const float* q11 = xn + (int)floorf(tf1) * kV + v1 + koffB * kColsPerN;
    const float* q21 = xn + (int)ceilf(tf1)  * kV + v1 + koffB * kColsPerN;

    // Batched gathers: 64 loads per k-chunk into distinct registers, issued
    // one chunk ahead so the WMMA sweep hides L2 latency.
    float xa0[16], xb0[16], xa1[16], xb1[16];
    #pragma unroll
    for (int h = 0; h < 16; ++h) {
      int off = h * kColsPerN;
      xa0[h] = q10[off]; xb0[h] = q20[off];
      xa1[h] = q11[off]; xb1[h] = q21[off];
    }

    #pragma unroll
    for (int kc = 0; kc < kC; kc += 32) {
      // Convert current batch (lerp + f16 pack) into B fragments.
      BFrag bf0, bf1;
      #pragma unroll
      for (int h = 0; h < 16; ++h) {
        bf0.h[h] = (_Float16)(a10 * xa0[h] + a20 * xb0[h]);
        bf1.h[h] = (_Float16)(a11 * xa1[h] + a21 * xb1[h]);
      }
      // Prefetch next k-chunk's 64 gathers (overlaps with the WMMAs below).
      if (kc + 32 < kC) {
        #pragma unroll
        for (int h = 0; h < 16; ++h) {
          int off = (kc + 32 + h) * kColsPerN;
          xa0[h] = q10[off]; xb0[h] = q20[off];
          xa1[h] = q11[off]; xb1[h] = q21[off];
        }
      }
      // Each A fragment (2x ds_load_b128) feeds two independent WMMAs.
      #pragma unroll
      for (int mt = 0; mt < 8; ++mt) {
        AFrag af;
        const _Float16* wp = &Wlds[(mt * 16 + mrow) * kC + kc + koffA];
        af.q[0] = *reinterpret_cast<const uint4*>(wp);
        af.q[1] = *reinterpret_cast<const uint4*>(wp + 8);
        acc[mt] = __builtin_amdgcn_wmma_f32_16x16x32_f16(
            false, af.v, false, bf0.v, (short)0, acc[mt], false, false);
        acc[8 + mt] = __builtin_amdgcn_wmma_f32_16x16x32_f16(
            false, af.v, false, bf1.v, (short)0, acc[8 + mt], false, false);
      }
    }
  }

  // Epilogue: C/D layout -> VGPR r holds M = r (lanes<16) / r+8 (lanes>=16).
  const int rowoff = (lane < 16) ? 0 : 8;
  float* op0 = out + (size_t)n * (kO * kColsPerN) + rem0;
  float* op1 = out + (size_t)n * (kO * kColsPerN) + rem1;
  #pragma unroll
  for (int mt = 0; mt < 8; ++mt) {
    #pragma unroll
    for (int r = 0; r < 8; ++r) {
      int o = mt * 16 + rowoff + r;
      float a = sA[o], b = sB[o];
      op0[o * kColsPerN] = acc[mt][r] * a + b;
      op1[o * kColsPerN] = acc[8 + mt][r] * a + b;
    }
  }
}

extern "C" void kernel_launch(void* const* d_in, const int* in_sizes, int n_in,
                              void* d_out, int out_size, void* d_ws, size_t ws_size,
                              hipStream_t stream) {
  const float* x  = (const float*)d_in[0];
  const float* tr = (const float*)d_in[1];
  const float* cw = (const float*)d_in[2];
  const float* cb = (const float*)d_in[3];
  const float* g  = (const float*)d_in[4];
  const float* be = (const float*)d_in[5];
  const float* mu = (const float*)d_in[6];
  const float* va = (const float*)d_in[7];
  float* out = (float*)d_out;

  const int use_tdm = (ws_size >= sizeof(_Float16) * (size_t)kWElems) ? 1 : 0;
  _Float16* wf16 = (_Float16*)d_ws;

  if (use_tdm) {
    convert_w_f16<<<(kWElems + 255) / 256, 256, 0, stream>>>(cw, wf16);
  }
  detgc_wmma_f16<<<kCols / kColsPerWG, 256, 0, stream>>>(
      x, tr, cw, cb, g, be, mu, va, wf16, use_tdm, out);
}
